// TrEnc_72653666779608
// MI455X (gfx1250) — matrix operand order repaired and checked by reading, not compiled, not run on hardware
//
#include <hip/hip_runtime.h>

typedef unsigned short u16;
typedef __attribute__((ext_vector_type(16))) __bf16 v16bf;
typedef __attribute__((ext_vector_type(8)))  float  v8f;
typedef __attribute__((ext_vector_type(4)))  float  v4f;
typedef __attribute__((ext_vector_type(4)))  unsigned uv4;

// ---- problem constants ----
#define NSEQ 8192
#define DMODEL 512
#define NHEAD 8
#define DHEAD 64
#define NF 266
#define NFP 288           // NF padded to a multiple of 32 (zero-padded)
#define D4 2048
#define DNORM  0.35355339059327373f   // 64^-0.25
#define DNORM2 0.125f                 // 64^-0.5
#define RATIO  0.0613139407475309f    // 266^-0.5
#define PEPS   1e-4f
#define LNEPS  1e-5f

// ---- CDNA5 async global->LDS copy (inline asm; device pass only) ----
__device__ __forceinline__ void async_copy16(const u16* g, u16* l) {
#ifdef __gfx1250__
    unsigned loff = (unsigned)(size_t)l;               // LDS aperture: addr[31:0] = LDS offset
    unsigned long long ga = (unsigned long long)(size_t)g;
    asm volatile("global_load_async_to_lds_b128 %0, %1, off"
                 :: "v"(loff), "v"(ga) : "memory");
#else
    *(uv4*)l = *(const uv4*)g;
#endif
}
__device__ __forceinline__ void async_wait0() {
#ifdef __gfx1250__
    asm volatile("s_wait_asynccnt 0x0" ::: "memory");
#endif
}

// ---- scalar helpers ----
__device__ __forceinline__ u16 f2bf(float f) {
    unsigned u = __float_as_uint(f);
    unsigned r = (u + 0x7FFFu + ((u >> 16) & 1u)) >> 16;
    return (u16)r;
}
__device__ __forceinline__ float bf2f(u16 h) {
    return __uint_as_float(((unsigned)h) << 16);
}
__device__ __forceinline__ unsigned fenc(float f) {
    unsigned u = __float_as_uint(f);
    return (u & 0x80000000u) ? ~u : (u | 0x80000000u);
}
__device__ __forceinline__ float fdec(unsigned e) {
    unsigned u = (e & 0x80000000u) ? (e & 0x7FFFFFFFu) : ~e;
    return __uint_as_float(u);
}
__device__ __forceinline__ float gelu_tanh(float x) {
    float x3 = x * x * x;
    return 0.5f * x * (1.0f + tanhf(0.7978845608028654f * (x + 0.044715f * x3)));
}
__device__ __forceinline__ float bredsum(float v, float* sh) {
    int t = threadIdx.x; sh[t] = v; __syncthreads();
    for (int s = 128; s; s >>= 1) { if (t < s) sh[t] += sh[t + s]; __syncthreads(); }
    float r = sh[0]; __syncthreads(); return r;
}
__device__ __forceinline__ float bredmax(float v, float* sh) {
    int t = threadIdx.x; sh[t] = v; __syncthreads();
    for (int s = 128; s; s >>= 1) { if (t < s) sh[t] = fmaxf(sh[t], sh[t + s]); __syncthreads(); }
    float r = sh[0]; __syncthreads(); return r;
}

// =====================================================================
// bf16 WMMA GEMM: C = act(alpha * A @ BT^T + bias) * rowscale + resid
//   A : [M,K]  row-major (lda), K-contiguous
//   BT: [N,K]  row-major (ldb), K-contiguous  (B pre-transposed)
// Tile 128x128x32, 256 threads = 8 waves (each 32x64 = 2x4 WMMA accums).
// Interior blocks: async global->LDS staging, double-buffered, no bounds
// logic in the loop. Edge blocks: guarded synchronous staging.
// =====================================================================
#define BM 128
#define BN 128
#define BK 32
#define SA 40   // padded LDS row stride (elements); 80B keeps 16B alignment

union FragU { v16bf v; v4f f[2]; };

__device__ __forceinline__ void wmma_step(const u16* sAb, const u16* sBb,
                                          v8f (&c)[2][4], int wm, int wn,
                                          int half, int l16)
{
    FragU a[2], b[4];
#pragma unroll
    for (int tm = 0; tm < 2; ++tm) {
        int m = wm * 32 + tm * 16 + l16;
        a[tm].f[0] = *(const v4f*)&sAb[m * SA + half * 8];       // k = 8h..8h+7
        a[tm].f[1] = *(const v4f*)&sAb[m * SA + 16 + half * 8];  // k = 16+8h..+7
    }
#pragma unroll
    for (int tn = 0; tn < 4; ++tn) {
        int n = wn * 64 + tn * 16 + l16;
        b[tn].f[0] = *(const v4f*)&sBb[n * SA + half * 8];       // k = 8h..8h+7
        b[tn].f[1] = *(const v4f*)&sBb[n * SA + 16 + half * 8];  // k = 16+8h..+7
    }
#pragma unroll
    for (int tm = 0; tm < 2; ++tm)
#pragma unroll
        for (int tn = 0; tn < 4; ++tn)
            c[tm][tn] = __builtin_amdgcn_wmma_f32_16x16x32_bf16(
                false, a[tm].v, false, b[tn].v, (short)0, c[tm][tn], false, false);
}

// guarded synchronous staging for edge blocks
__device__ __forceinline__ void stage_guarded(const u16* __restrict__ src, int ld,
                                              int row0, int rowsMax, int k0, u16* sbuf)
{
#pragma unroll
    for (int i = 0; i < 2; ++i) {
        int cidx = threadIdx.x + i * 256;      // 512 chunks of 8 elems (16B)
        int r  = cidx >> 2;
        int kc = (cidx & 3) * 8;
        int gr = row0 + r;
        int cg = gr < rowsMax ? gr : rowsMax - 1;      // clamped, branchless
        uv4 d = *(const uv4*)(src + (size_t)cg * ld + k0 + kc);
        if (gr >= rowsMax) { uv4 z = {0u, 0u, 0u, 0u}; d = z; }
        *(uv4*)&sbuf[r * SA + kc] = d;
    }
}

__global__ __launch_bounds__(256)
void k_gemm_bf16(const u16* __restrict__ A, int lda,
                 const u16* __restrict__ BT, int ldb,
                 float* __restrict__ Cf, u16* __restrict__ Cbf, int ldc, int transC,
                 const float* __restrict__ bias,
                 const float* __restrict__ resid, int ldr,
                 const float* __restrict__ rowscale,
                 int M, int Nn, int K, float alpha, int act)
{
    __shared__ __align__(16) u16 sA[2][BM * SA];
    __shared__ __align__(16) u16 sB[2][BN * SA];

    const int tid  = threadIdx.x;
    const int lane = tid & 31;
    const int wave = tid >> 5;
    const int wm   = wave & 3;          // 4 waves along M (32 rows each)
    const int wn   = wave >> 2;         // 2 waves along N (64 cols each)
    const int m0   = blockIdx.y * BM;
    const int n0   = blockIdx.x * BN;
    const int half = lane >> 4;
    const int l16  = lane & 15;

    v8f c[2][4];
#pragma unroll
    for (int tm = 0; tm < 2; ++tm)
#pragma unroll
        for (int tn = 0; tn < 4; ++tn)
#pragma unroll
            for (int r = 0; r < 8; ++r) c[tm][tn][r] = 0.0f;

    if ((m0 + BM <= M) && (n0 + BN <= Nn)) {
        // ---------- FAST: interior block, async staging, zero bounds logic ----------
        const int c0  = tid,        c1  = tid + 256;
        const int r0c = c0 >> 2,    k0c = (c0 & 3) * 8;
        const int r1c = c1 >> 2,    k1c = (c1 & 3) * 8;
        const u16* pA0 = A  + (size_t)(m0 + r0c) * lda + k0c;
        const u16* pA1 = A  + (size_t)(m0 + r1c) * lda + k1c;
        const u16* pB0 = BT + (size_t)(n0 + r0c) * ldb + k0c;
        const u16* pB1 = BT + (size_t)(n0 + r1c) * ldb + k1c;
        u16* lA0[2] = { &sA[0][r0c * SA + k0c], &sA[1][r0c * SA + k0c] };
        u16* lA1[2] = { &sA[0][r1c * SA + k1c], &sA[1][r1c * SA + k1c] };
        u16* lB0[2] = { &sB[0][r0c * SA + k0c], &sB[1][r0c * SA + k0c] };
        u16* lB1[2] = { &sB[0][r1c * SA + k1c], &sB[1][r1c * SA + k1c] };

        async_copy16(pA0, lA0[0]); async_copy16(pA1, lA1[0]);
        async_copy16(pB0, lB0[0]); async_copy16(pB1, lB1[0]);
        async_wait0();
        __syncthreads();

        int buf = 0;
        for (int k0 = 0; k0 < K; k0 += BK) {
            if (k0 + BK < K) {
                int kn = k0 + BK;
                async_copy16(pA0 + kn, lA0[buf ^ 1]); async_copy16(pA1 + kn, lA1[buf ^ 1]);
                async_copy16(pB0 + kn, lB0[buf ^ 1]); async_copy16(pB1 + kn, lB1[buf ^ 1]);
            }
            wmma_step(sA[buf], sB[buf], c, wm, wn, half, l16);
            async_wait0();
            __syncthreads();
            buf ^= 1;
        }
    } else {
        // ---------- GENERIC: edge block, guarded synchronous staging ----------
        stage_guarded(A, lda, m0, M, 0, sA[0]);
        stage_guarded(BT, ldb, n0, Nn, 0, sB[0]);
        __syncthreads();
        int buf = 0;
        for (int k0 = 0; k0 < K; k0 += BK) {
            if (k0 + BK < K) {
                stage_guarded(A, lda, m0, M, k0 + BK, sA[buf ^ 1]);
                stage_guarded(BT, ldb, n0, Nn, k0 + BK, sB[buf ^ 1]);
            }
            wmma_step(sA[buf], sB[buf], c, wm, wn, half, l16);
            __syncthreads();
            buf ^= 1;
        }
    }

    // ---- epilogue ----
#pragma unroll
    for (int tm = 0; tm < 2; ++tm)
#pragma unroll
        for (int tn = 0; tn < 4; ++tn)
#pragma unroll
            for (int r = 0; r < 8; ++r) {
                int gm = m0 + wm * 32 + tm * 16 + half * 8 + r;
                int gn = n0 + wn * 64 + tn * 16 + l16;
                if (gm < M && gn < Nn) {
                    float v = c[tm][tn][r] * alpha;
                    if (bias)     v += bias[gn];
                    if (act == 1) v = gelu_tanh(v);
                    if (rowscale) v *= rowscale[gm];
                    if (resid)    v += resid[(size_t)gm * ldr + gn];
                    if (Cf)  Cf[(size_t)gm * ldc + gn] = v;
                    if (Cbf) {
                        if (transC) Cbf[(size_t)gn * ldc + gm] = f2bf(v);
                        else        Cbf[(size_t)gm * ldc + gn] = f2bf(v);
                    }
                }
            }
}

// ---------------- LayerNorm (D=512), fp32 in -> bf16 out ----------------
__global__ __launch_bounds__(256)
void k_ln(const float* __restrict__ x, const float* __restrict__ g,
          const float* __restrict__ b, u16* __restrict__ out)
{
    __shared__ float sh[256];
    int row = blockIdx.x, t = threadIdx.x;
    const float* xr = x + (size_t)row * DMODEL;
    float v0 = xr[t], v1 = xr[t + 256];
    float mu = bredsum(v0 + v1, sh) * (1.0f / DMODEL);
    float d0 = v0 - mu, d1 = v1 - mu;
    float var = bredsum(d0 * d0 + d1 * d1, sh) * (1.0f / DMODEL);
    float rs = rsqrtf(var + LNEPS);
    out[(size_t)row * DMODEL + t]       = f2bf(d0 * rs * g[t] + b[t]);
    out[(size_t)row * DMODEL + t + 256] = f2bf(d1 * rs * g[t + 256] + b[t + 256]);
}

// ---- fp32 [K][Nc] -> bf16 transposed [Nc][K] (tiled) ----
__global__ __launch_bounds__(256)
void k_f2bfT(const float* __restrict__ in, u16* __restrict__ out, int K, int Nc)
{
    __shared__ float tile[32][33];
    int tx = threadIdx.x & 31, ty = threadIdx.x >> 5;
    int k0 = blockIdx.y * 32, n0 = blockIdx.x * 32;
#pragma unroll
    for (int i = 0; i < 4; ++i) {
        int k = k0 + ty + i * 8, n = n0 + tx;
        tile[ty + i * 8][tx] = (k < K && n < Nc) ? in[(size_t)k * Nc + n] : 0.0f;
    }
    __syncthreads();
#pragma unroll
    for (int i = 0; i < 4; ++i) {
        int n = n0 + ty + i * 8, k = k0 + tx;
        if (n < Nc && k < K) out[(size_t)n * K + k] = f2bf(tile[tx][ty + i * 8]);
    }
}

// ---- bf16 [R][C] -> [C][R] transpose (tiled, z-batched) ----
__global__ __launch_bounds__(256)
void k_trbf(const u16* __restrict__ in, u16* __restrict__ out, int R, int C,
            long inStride, long outStride)
{
    __shared__ u16 tile[32][33];
    const u16* src = in + (size_t)blockIdx.z * inStride;
    u16* dst = out + (size_t)blockIdx.z * outStride;
    int tx = threadIdx.x & 31, ty = threadIdx.x >> 5;
    int c0 = blockIdx.x * 32, r0 = blockIdx.y * 32;
#pragma unroll
    for (int i = 0; i < 4; ++i) {
        int r = r0 + ty + i * 8, cc = c0 + tx;
        tile[ty + i * 8][tx] = (r < R && cc < C) ? src[(size_t)r * C + cc] : (u16)0;
    }
    __syncthreads();
#pragma unroll
    for (int i = 0; i < 4; ++i) {
        int cc = c0 + ty + i * 8, r = r0 + tx;
        if (cc < C && r < R) dst[(size_t)cc * R + r] = tile[tx][ty + i * 8];
    }
}

// ---- proj [NF,64] -> padded bf16 [NFP,64] (already BT layout) ----
__global__ void k_projP(const float* __restrict__ proj, u16* __restrict__ out)
{
    int n = NFP * DHEAD;
    for (int i = blockIdx.x * blockDim.x + threadIdx.x; i < n; i += gridDim.x * blockDim.x) {
        int j = i / DHEAD, k = i % DHEAD;
        out[i] = (j < NF) ? f2bf(proj[(size_t)j * DHEAD + k]) : (u16)0;
    }
}

__global__ void k_zero(unsigned* __restrict__ p, int n)
{
    for (int i = blockIdx.x * blockDim.x + threadIdx.x; i < n; i += gridDim.x * blockDim.x)
        p[i] = 0u;
}

// ---- per-head global max over dd_k (monotone-encoded atomicMax) ----
__global__ __launch_bounds__(256)
void k_kmax(const float* __restrict__ dd, unsigned* __restrict__ kmax_enc)
{
    __shared__ float sh[256];
    int h = blockIdx.y, t = threadIdx.x;
    long long total = (long long)NSEQ * NF;
    float lm = -3.4e38f;
    for (long long i = (long long)blockIdx.x * 256 + t; i < total; i += (long long)gridDim.x * 256) {
        int n = (int)(i / NF), j = (int)(i % NF);
        lm = fmaxf(lm, dd[((size_t)h * NSEQ + n) * NFP + j]);
    }
    float bm = bredmax(lm, sh);
    if (t == 0) atomicMax(kmax_enc + h, fenc(bm));
}

// ---- phi: qp/kp = ratio*(exp(dd - diag - m) + eps), zero padded ----
__global__ __launch_bounds__(256)
void k_phi(const float* __restrict__ dd, const u16* __restrict__ qk,
           const unsigned* __restrict__ kmax_enc, u16* __restrict__ out, int is_query)
{
    __shared__ float sh[256];
    int r = blockIdx.x;             // r = h*NSEQ + n
    int h = r >> 13;
    int n = r & (NSEQ - 1);
    int t = threadIdx.x;

    float qv = 0.0f;
    if (t < DHEAD) {
        float q = bf2f(qk[(size_t)n * DMODEL + h * DHEAD + t]);
        qv = q * q;
    }
    float diag = bredsum(qv, sh) * 0.5f * DNORM2;

    const float* ddr = dd + (size_t)r * NFP;
    float m;
    if (is_query) {
        float lm = -3.4e38f;
        for (int j = t; j < NF; j += 256) lm = fmaxf(lm, ddr[j]);
        m = bredmax(lm, sh);
    } else {
        m = fdec(kmax_enc[h]);
    }
    u16* outr = out + (size_t)r * NFP;
    for (int j = t; j < NFP; j += 256) {
        float v = 0.0f;
        if (j < NF) v = RATIO * (__expf(ddr[j] - diag - m) + PEPS);
        outr[j] = f2bf(v);
    }
}

// ---- kp_sum[h][j] = sum_n kpT[h][j][n]  (coalesced over n) ----
__global__ __launch_bounds__(256)
void k_kpsum(const u16* __restrict__ kpT, float* __restrict__ kp_sum)
{
    __shared__ float sh[256];
    int h = blockIdx.x / NFP, j = blockIdx.x % NFP, t = threadIdx.x;
    const u16* p = kpT + ((size_t)h * NFP + j) * NSEQ;
    float s = 0.0f;
    for (int n = t; n < NSEQ; n += 256) s += bf2f(p[n]);
    float tot = bredsum(s, sh);
    if (t == 0) kp_sum[h * NFP + j] = tot;
}

// ---- d_inv[h][n] = 1 / (qp[h][n] . kp_sum[h]) ----
__global__ __launch_bounds__(256)
void k_dinv(const u16* __restrict__ qp, const float* __restrict__ kp_sum,
            float* __restrict__ d_inv)
{
    __shared__ float sh[256];
    int r = blockIdx.x;
    int h = r >> 13;
    int t = threadIdx.x;
    float s = 0.0f;
    for (int j = t; j < NFP; j += 256)
        s += bf2f(qp[(size_t)r * NFP + j]) * kp_sum[h * NFP + j];
    float tot = bredsum(s, sh);
    if (t == 0) d_inv[r] = 1.0f / tot;
}

// =====================================================================
extern "C" void kernel_launch(void* const* d_in, const int* in_sizes, int n_in,
                              void* d_out, int out_size, void* d_ws, size_t ws_size,
                              hipStream_t stream)
{
    (void)in_sizes; (void)n_in; (void)out_size; (void)ws_size;
    const float* x     = (const float*)d_in[0];
    const float* proj  = (const float*)d_in[1];
    const float* ln1_g = (const float*)d_in[2];
    const float* ln1_b = (const float*)d_in[3];
    const float* wq    = (const float*)d_in[4];
    const float* wk    = (const float*)d_in[5];
    const float* wv    = (const float*)d_in[6];
    const float* wo    = (const float*)d_in[7];
    const float* wo_b  = (const float*)d_in[8];
    const float* ln2_g = (const float*)d_in[9];
    const float* ln2_b = (const float*)d_in[10];
    const float* w1    = (const float*)d_in[11];
    const float* b1    = (const float*)d_in[12];
    const float* w2    = (const float*)d_in[13];
    const float* b2    = (const float*)d_in[14];
    float* out = (float*)d_out;

    // ---- workspace bump allocator ----
    char* ws = (char*)d_ws;
    size_t off = 0;
    auto alc = [&](size_t bytes) -> void* {
        off = (off + 255) & ~(size_t)255;
        void* p = ws + off; off += bytes; return p;
    };
    u16* hbf   = (u16*)alc((size_t)NSEQ * DMODEL * 2);
    u16* qbf   = (u16*)alc((size_t)NSEQ * DMODEL * 2);
    u16* kbf   = (u16*)alc((size_t)NSEQ * DMODEL * 2);
    u16* vbf   = (u16*)alc((size_t)NSEQ * DMODEL * 2);
    u16* vT    = (u16*)alc((size_t)DMODEL * NSEQ * 2);          // [512][8192]
    u16* wqT   = (u16*)alc((size_t)DMODEL * DMODEL * 2);
    u16* wkT   = (u16*)alc((size_t)DMODEL * DMODEL * 2);
    u16* wvT   = (u16*)alc((size_t)DMODEL * DMODEL * 2);
    u16* woT   = (u16*)alc((size_t)DMODEL * DMODEL * 2);
    u16* w1T   = (u16*)alc((size_t)D4 * DMODEL * 2);            // [2048][512]
    u16* w2T   = (u16*)alc((size_t)DMODEL * D4 * 2);            // [512][2048]
    u16* projP = (u16*)alc((size_t)NFP * DHEAD * 2);            // [288][64]
    float* dd_q = (float*)alc((size_t)NHEAD * NSEQ * NFP * 4);
    float* dd_k = (float*)alc((size_t)NHEAD * NSEQ * NFP * 4);
    u16* qp    = (u16*)alc((size_t)NHEAD * NSEQ * NFP * 2);
    u16* kp    = (u16*)alc((size_t)NHEAD * NSEQ * NFP * 2);
    u16* kpT   = (u16*)alc((size_t)NHEAD * NFP * NSEQ * 2);     // [h][288][8192]
    unsigned* kmax_enc = (unsigned*)alc(NHEAD * 4);
    float* kp_sum = (float*)alc((size_t)NHEAD * NFP * 4);
    float* d_inv  = (float*)alc((size_t)NHEAD * NSEQ * 4);
    u16* ctxT  = (u16*)alc((size_t)NHEAD * DHEAD * NFP * 2);    // [h][64][288]
    u16* attn_bf = (u16*)alc((size_t)NSEQ * DMODEL * 2);
    float* x2  = (float*)alc((size_t)NSEQ * DMODEL * 4);
    u16* h2bf  = (u16*)alc((size_t)NSEQ * DMODEL * 2);
    u16* y1bf  = (u16*)alc((size_t)NSEQ * D4 * 2);

    auto gemm = [&](const u16* A, int lda, const u16* BT, int ldb,
                    float* Cf, u16* Cbf, int ldc, int transC,
                    const float* bias, const float* resid, int ldr,
                    const float* rowscale, int M, int Nn, int K, float alpha, int act) {
        dim3 g((Nn + BN - 1) / BN, (M + BM - 1) / BM);
        k_gemm_bf16<<<g, 256, 0, stream>>>(A, lda, BT, ldb, Cf, Cbf, ldc, transC,
                                           bias, resid, ldr, rowscale, M, Nn, K, alpha, act);
    };

    // ---- weight conversions (transposed, K-contiguous) ----
    k_f2bfT<<<dim3(16, 16), 256, 0, stream>>>(wq, wqT, DMODEL, DMODEL);
    k_f2bfT<<<dim3(16, 16), 256, 0, stream>>>(wk, wkT, DMODEL, DMODEL);
    k_f2bfT<<<dim3(16, 16), 256, 0, stream>>>(wv, wvT, DMODEL, DMODEL);
    k_f2bfT<<<dim3(16, 16), 256, 0, stream>>>(wo, woT, DMODEL, DMODEL);
    k_f2bfT<<<dim3(64, 16), 256, 0, stream>>>(w1, w1T, DMODEL, D4);
    k_f2bfT<<<dim3(16, 64), 256, 0, stream>>>(w2, w2T, D4, DMODEL);
    k_projP<<<72, 256, 0, stream>>>(proj, projP);

    // ---- LN1, QKV ----
    k_ln<<<NSEQ, 256, 0, stream>>>(x, ln1_g, ln1_b, hbf);
    gemm(hbf, DMODEL, wqT, DMODEL, nullptr, qbf, DMODEL, 0,
         nullptr, nullptr, 0, nullptr, NSEQ, DMODEL, DMODEL, 1.0f, 0);
    gemm(hbf, DMODEL, wkT, DMODEL, nullptr, kbf, DMODEL, 0,
         nullptr, nullptr, 0, nullptr, NSEQ, DMODEL, DMODEL, 1.0f, 0);
    gemm(hbf, DMODEL, wvT, DMODEL, nullptr, vbf, DMODEL, 0,
         nullptr, nullptr, 0, nullptr, NSEQ, DMODEL, DMODEL, 1.0f, 0);

    // ---- dd = dnorm * (q|k head) @ proj^T  [per head] ----
    for (int h = 0; h < NHEAD; ++h) {
        gemm(qbf + h * DHEAD, DMODEL, projP, DHEAD,
             dd_q + (size_t)h * NSEQ * NFP, nullptr, NFP, 0,
             nullptr, nullptr, 0, nullptr, NSEQ, NFP, DHEAD, DNORM, 0);
        gemm(kbf + h * DHEAD, DMODEL, projP, DHEAD,
             dd_k + (size_t)h * NSEQ * NFP, nullptr, NFP, 0,
             nullptr, nullptr, 0, nullptr, NSEQ, NFP, DHEAD, DNORM, 0);
    }

    // ---- phi(q), phi(k) ----
    k_zero<<<1, 256, 0, stream>>>(kmax_enc, NHEAD);
    {
        dim3 g(512, NHEAD);
        k_kmax<<<g, 256, 0, stream>>>(dd_k, kmax_enc);
    }
    k_phi<<<NHEAD * NSEQ, 256, 0, stream>>>(dd_q, qbf, kmax_enc, qp, 1);
    k_phi<<<NHEAD * NSEQ, 256, 0, stream>>>(dd_k, kbf, kmax_enc, kp, 0);

    // ---- transposes: v -> vT, kp -> kpT ----
    k_trbf<<<dim3(16, 256, 1), 256, 0, stream>>>(vbf, vT, NSEQ, DMODEL, 0, 0);
    k_trbf<<<dim3(9, 256, NHEAD), 256, 0, stream>>>(kp, kpT, NSEQ, NFP,
                                                    (long)NSEQ * NFP, (long)NFP * NSEQ);
    k_kpsum<<<NHEAD * NFP, 256, 0, stream>>>(kpT, kp_sum);
    k_dinv<<<NHEAD * NSEQ, 256, 0, stream>>>(qp, kp_sum, d_inv);

    // ---- context = kp^T @ v (write ctxT); attn = (qp @ ctx) * d_inv ----
    for (int h = 0; h < NHEAD; ++h) {
        gemm(kpT + (size_t)h * NFP * NSEQ, NSEQ,           // A = kp^T [NFP][N]
             vT + (size_t)h * DHEAD * NSEQ, NSEQ,          // BT = v^T head [64][N]
             nullptr, ctxT + (size_t)h * DHEAD * NFP, NFP, 1,
             nullptr, nullptr, 0, nullptr, NFP, DHEAD, NSEQ, 1.0f, 0);
        gemm(qp + (size_t)h * NSEQ * NFP, NFP,             // A = qp [N][NFP]
             ctxT + (size_t)h * DHEAD * NFP, NFP,          // BT = ctx^T [64][NFP]
             nullptr, attn_bf + h * DHEAD, DMODEL, 0,
             nullptr, nullptr, 0, d_inv + (size_t)h * NSEQ, NSEQ, DHEAD, NFP, 1.0f, 0);
    }

    // ---- x2 = x + attn @ wo + wo_b ----
    gemm(attn_bf, DMODEL, woT, DMODEL, x2, nullptr, DMODEL, 0,
         wo_b, x, DMODEL, nullptr, NSEQ, DMODEL, DMODEL, 1.0f, 0);

    // ---- FFN: out = x2 + gelu(LN2(x2) @ w1 + b1) @ w2 + b2 ----
    k_ln<<<NSEQ, 256, 0, stream>>>(x2, ln2_g, ln2_b, h2bf);
    gemm(h2bf, DMODEL, w1T, DMODEL, nullptr, y1bf, D4, 0,
         b1, nullptr, 0, nullptr, NSEQ, D4, DMODEL, 1.0f, 1);
    gemm(y1bf, D4, w2T, D4, out, nullptr, DMODEL, 0,
         b2, x2, DMODEL, nullptr, NSEQ, DMODEL, D4, 1.0f, 0);
}
